// GATNet_8426725835195
// MI455X (gfx1250) — matrix operand order repaired and checked by reading, not compiled, not run on hardware
//
#include <hip/hip_runtime.h>
#include <hip/hip_bf16.h>
#include <math.h>

// ---------------- problem constants (match reference) ----------------
#define NNODE 10000
#define NEDGE 160000
#define ETOT  170000          // NEDGE + NNODE self loops
#define FIN   78
#define KP1   96              // FIN padded to multiple of 32
#define HEADS 10
#define CCH   128             // channels per head
#define HC    1280            // HEADS * CCH
#define NGRAPH 64

typedef __attribute__((ext_vector_type(16))) _Float16 v16h;
typedef __attribute__((ext_vector_type(8)))  float    v8f;

// ---------------- ordered-float <-> uint mapping for atomic max ------
__device__ __forceinline__ unsigned int fmap(float f) {
    unsigned int u = __float_as_uint(f);
    return (u & 0x80000000u) ? ~u : (u | 0x80000000u);
}
__device__ __forceinline__ float funmap(unsigned int u) {
    return (u & 0x80000000u) ? __uint_as_float(u & 0x7fffffffu)
                             : __uint_as_float(~u);
}
__device__ __forceinline__ void atomicMaxF(unsigned int* p, float v) {
    atomicMax(p, fmap(v));
}

// ---------------- generic fills --------------------------------------
__global__ void fill_f32(float* p, int n, float v) {
    int i = blockIdx.x * 256 + threadIdx.x;
    if (i < n) p[i] = v;
}
__global__ void fill_u32(unsigned int* p, int n, unsigned int v) {
    int i = blockIdx.x * 256 + threadIdx.x;
    if (i < n) p[i] = v;
}

// ---------------- f32 -> f16 cast / pad / transpose ------------------
__global__ void cast_pad_x(const float* __restrict__ x, _Float16* __restrict__ xh) {
    int i = blockIdx.x * 256 + threadIdx.x;            // over NNODE*KP1
    if (i >= NNODE * KP1) return;
    int n = i / KP1, k = i - n * KP1;
    xh[i] = (k < FIN) ? (_Float16)x[n * FIN + k] : (_Float16)0.0f;
}
// W1: [FIN][HC] -> w1t: [HC][KP1]  (transposed, zero padded K)
__global__ void tcast_w1(const float* __restrict__ W1, _Float16* __restrict__ w1t) {
    int i = blockIdx.x * 256 + threadIdx.x;            // over HC*KP1
    if (i >= HC * KP1) return;
    int nn = i / KP1, k = i - nn * KP1;
    w1t[i] = (k < FIN) ? (_Float16)W1[k * HC + nn] : (_Float16)0.0f;
}
// W2: [HC][CCH] -> w2t: [CCH][HC]
__global__ void tcast_w2(const float* __restrict__ W2, _Float16* __restrict__ w2t) {
    int i = blockIdx.x * 256 + threadIdx.x;            // over CCH*HC
    if (i >= CCH * HC) return;
    int nn = i / HC, k = i - nn * HC;
    w2t[i] = (_Float16)W2[k * CCH + nn];
}

// ---------------- WMMA GEMM: C[M][N] = A[M][K] * Bt[N][K]^T ----------
// One wave per 16x64 output strip (4 N-tiles): A fragment loaded once per
// K-step and reused across 4 back-to-back v_wmma_f32_16x16x32_f16.
// A fragment: per ISA "16-bit A-Matrix 16x32" table (two b128 loads).
// B fragment: same K order as A fragment (ISA table order, two b128 loads).
__device__ __forceinline__ v16h load_fragA(const _Float16* __restrict__ A,
                                           int ld, int row, int kb, int half) {
    union { v16h v; uint4 q[2]; } f;
    const uint4* p = reinterpret_cast<const uint4*>(A + (size_t)row * ld + kb + half * 8);
    f.q[0] = p[0];          // K pairs {0,2,4,6}+half*8
    f.q[1] = p[2];          // K pairs {16,18,20,22}+half*8
    return f.v;
}
__device__ __forceinline__ v16h load_fragB(const _Float16* __restrict__ Bt,
                                           int ld, int row, int kb, int half) {
    union { v16h v; uint4 q[2]; } f;
    const uint4* p = reinterpret_cast<const uint4*>(Bt + (size_t)row * ld + kb + half * 8);
    f.q[0] = p[0];
    f.q[1] = p[2];
    return f.v;
}

#define WMMA32(a, b, c) __builtin_amdgcn_wmma_f32_16x16x32_f16( \
        false, (a), false, (b), (short)0, (c), false, false)

__global__ void wmma_gemm_f16(const _Float16* __restrict__ A,
                              const _Float16* __restrict__ Bt,
                              float* __restrict__ C,
                              int M, int Nn, int K) {
    int wid  = blockIdx.x * (blockDim.x >> 5) + (threadIdx.x >> 5);
    int lane = threadIdx.x & 31;
    int tn4 = Nn >> 6;                        // groups of 4 n-tiles (N % 64 == 0)
    int groups = (M >> 4) * tn4;
    if (wid >= groups) return;                // wave-uniform: EXEC stays all-ones
    int tmi = wid / tn4, tg = wid - tmi * tn4;
    int half = lane >> 4, r = lane & 15;
    int mrow  = tmi * 16 + r;                 // A row for this lane
    int nbase = tg * 64;                      // first output column of this strip
    v8f acc0 = {}, acc1 = {}, acc2 = {}, acc3 = {};
    for (int kb = 0; kb < K; kb += 32) {
        if (kb + 32 < K) {
            __builtin_prefetch(A  + (size_t)mrow * K + kb + 32, 0, 1);
            __builtin_prefetch(Bt + (size_t)(nbase + r) * K + kb + 32, 0, 1);
        }
        v16h a  = load_fragA(A,  K, mrow, kb, half);
        v16h b0 = load_fragB(Bt, K, nbase +  0 + r, kb, half);
        v16h b1 = load_fragB(Bt, K, nbase + 16 + r, kb, half);
        v16h b2 = load_fragB(Bt, K, nbase + 32 + r, kb, half);
        v16h b3 = load_fragB(Bt, K, nbase + 48 + r, kb, half);
        acc0 = WMMA32(a, b0, acc0);           // 4 independent WMMAs: no D->A/B
        acc1 = WMMA32(a, b1, acc1);           // hazard, pipeline back-to-back
        acc2 = WMMA32(a, b2, acc2);
        acc3 = WMMA32(a, b3, acc3);
    }
    int m0 = tmi * 16 + half * 8;
    union { v8f v; float f[8]; } o;
    o.v = acc0;
    #pragma unroll
    for (int i = 0; i < 8; ++i) C[(size_t)(m0 + i) * Nn + nbase +  0 + r] = o.f[i];
    o.v = acc1;
    #pragma unroll
    for (int i = 0; i < 8; ++i) C[(size_t)(m0 + i) * Nn + nbase + 16 + r] = o.f[i];
    o.v = acc2;
    #pragma unroll
    for (int i = 0; i < 8; ++i) C[(size_t)(m0 + i) * Nn + nbase + 32 + r] = o.f[i];
    o.v = acc3;
    #pragma unroll
    for (int i = 0; i < 8; ++i) C[(size_t)(m0 + i) * Nn + nbase + 48 + r] = o.f[i];
}

// ---------------- attention coefficient dots -------------------------
// out[n][h] = sum_c h[n][h*CCH+c] * av[h*CCH+c]
__global__ void attn_dot(const float* __restrict__ h, const float* __restrict__ av,
                         float* __restrict__ out, int heads) {
    int idx = blockIdx.x * 256 + threadIdx.x;
    if (idx >= NNODE * heads) return;
    int n = idx / heads, hh = idx - n * heads;
    const float* hp = h + (size_t)(n * heads + hh) * CCH;
    const float* ap = av + hh * CCH;
    float s = 0.0f;
    #pragma unroll 8
    for (int c = 0; c < CCH; ++c) s += hp[c] * ap[c];
    out[idx] = s;
}

// ---------------- edge softmax pipeline ------------------------------
__device__ __forceinline__ void edge_sd(const int* ei, int eid, int& s, int& d) {
    if (eid < NEDGE) { s = ei[eid]; d = ei[NEDGE + eid]; }
    else             { s = d = eid - NEDGE; }
}

__global__ void edge_score(const int* __restrict__ ei,
                           const float* __restrict__ asrc, const float* __restrict__ adst,
                           float* __restrict__ e, unsigned int* __restrict__ m, int heads) {
    int idx = blockIdx.x * 256 + threadIdx.x;
    if (idx >= ETOT * heads) return;
    int eid = idx / heads, hh = idx - eid * heads;
    int s, d; edge_sd(ei, eid, s, d);
    float t = asrc[s * heads + hh] + adst[d * heads + hh];
    t = (t > 0.0f) ? t : 0.2f * t;            // leaky_relu(0.2)
    e[idx] = t;
    atomicMaxF(m + d * heads + hh, t);
}

__global__ void decode_m(unsigned int* m, int n) {
    int i = blockIdx.x * 256 + threadIdx.x;
    if (i >= n) return;
    float f = funmap(m[i]);
    if (!(f > -INFINITY && f < INFINITY)) f = 0.0f;   // isfinite replacement
    m[i] = __float_as_uint(f);
}

__global__ void edge_exp(const int* __restrict__ ei, float* __restrict__ e,
                         const float* __restrict__ mf, float* __restrict__ den, int heads) {
    int idx = blockIdx.x * 256 + threadIdx.x;
    if (idx >= ETOT * heads) return;
    int eid = idx / heads, hh = idx - eid * heads;
    int s, d; edge_sd(ei, eid, s, d);
    float ex = expf(e[idx] - mf[d * heads + hh]);
    e[idx] = ex;
    atomicAdd(den + d * heads + hh, ex);
}

__global__ void edge_alpha(const int* __restrict__ ei, const float* __restrict__ e,
                           const float* __restrict__ den, float* __restrict__ out, int heads) {
    int idx = blockIdx.x * 256 + threadIdx.x;
    if (idx >= ETOT * heads) return;
    int eid = idx / heads, hh = idx - eid * heads;
    int s, d; edge_sd(ei, eid, s, d);
    (void)s;
    out[idx] = e[idx] / (den[d * heads + hh] + 1e-16f);
}

// one block (128 threads) per (edge, head); thread = channel
__global__ void aggregate(const int* __restrict__ ei, const float* __restrict__ h,
                          const float* __restrict__ alpha, float* __restrict__ out, int heads) {
    int gid = blockIdx.x;                     // (eid*heads + hh)
    int eid = gid / heads, hh = gid - eid * heads;
    int s, d; edge_sd(ei, eid, s, d);
    float a = alpha[gid];
    int c = threadIdx.x;                      // 0..127
    int off = hh * CCH + c;
    atomicAdd(out + (size_t)d * heads * CCH + off,
              h[(size_t)s * heads * CCH + off] * a);
}

// ---------------- elu + cast -----------------------------------------
__global__ void elu_cast1(const float* __restrict__ out1, const float* __restrict__ b1,
                          _Float16* __restrict__ h1h) {
    int i = blockIdx.x * 256 + threadIdx.x;
    if (i >= NNODE * HC) return;
    float v = out1[i] + b1[i % HC];
    v = (v > 0.0f) ? v : expm1f(v);
    h1h[i] = (_Float16)v;
}
__global__ void elu2_inplace(float* __restrict__ out2, const float* __restrict__ b2) {
    int i = blockIdx.x * 256 + threadIdx.x;
    if (i >= NNODE * CCH) return;
    float v = out2[i] + b2[i % CCH];
    out2[i] = (v > 0.0f) ? v : expm1f(v);
}

// ---------------- global max pool ------------------------------------
__global__ void pool_max(const float* __restrict__ h2, const int* __restrict__ batch,
                         unsigned int* __restrict__ pm) {
    int i = blockIdx.x * 256 + threadIdx.x;
    if (i >= NNODE * CCH) return;
    int n = i / CCH, c = i - n * CCH;
    atomicMaxF(pm + batch[n] * CCH + c, h2[i]);
}
__global__ void decode_pool(const unsigned int* __restrict__ pm, float* __restrict__ out) {
    int i = blockIdx.x * 256 + threadIdx.x;
    if (i >= NGRAPH * CCH) return;
    out[i] = funmap(pm[i]);                   // empty graph -> -inf (matches segment_max)
}

// ---------------- launch ---------------------------------------------
extern "C" void kernel_launch(void* const* d_in, const int* in_sizes, int n_in,
                              void* d_out, int out_size, void* d_ws, size_t ws_size,
                              hipStream_t stream) {
    const float* x   = (const float*)d_in[0];
    const int*   ei  = (const int*)  d_in[1];
    const int*   bat = (const int*)  d_in[2];
    const float* W1  = (const float*)d_in[3];
    const float* as1 = (const float*)d_in[4];
    const float* ad1 = (const float*)d_in[5];
    const float* b1  = (const float*)d_in[6];
    const float* W2  = (const float*)d_in[7];
    const float* as2 = (const float*)d_in[8];
    const float* ad2 = (const float*)d_in[9];
    const float* b2  = (const float*)d_in[10];
    float* out = (float*)d_out;
    float* alpha1_out = out + NGRAPH * CCH;   // second tuple element

    // bump allocator over workspace (256B aligned)
    size_t off = 0;
    auto alloc = [&](size_t bytes) -> void* {
        void* p = (char*)d_ws + off;
        off = (off + bytes + 255) & ~(size_t)255;
        return p;
    };
    _Float16* xh   = (_Float16*)alloc((size_t)NNODE * KP1 * 2);
    _Float16* w1t  = (_Float16*)alloc((size_t)HC * KP1 * 2);
    float*    h1   = (float*)   alloc((size_t)NNODE * HC * 4);
    float*    asr1 = (float*)   alloc((size_t)NNODE * HEADS * 4);
    float*    ads1 = (float*)   alloc((size_t)NNODE * HEADS * 4);
    float*    e1   = (float*)   alloc((size_t)ETOT * HEADS * 4);
    unsigned int* m1 = (unsigned int*)alloc((size_t)NNODE * HEADS * 4);
    float*    den1 = (float*)   alloc((size_t)NNODE * HEADS * 4);
    float*    out1 = (float*)   alloc((size_t)NNODE * HC * 4);
    _Float16* h1h  = (_Float16*)alloc((size_t)NNODE * HC * 2);
    _Float16* w2t  = (_Float16*)alloc((size_t)CCH * HC * 2);
    float*    h2   = (float*)   alloc((size_t)NNODE * CCH * 4);
    float*    asr2 = (float*)   alloc((size_t)NNODE * 4);
    float*    ads2 = (float*)   alloc((size_t)NNODE * 4);
    float*    e2   = (float*)   alloc((size_t)ETOT * 4);
    unsigned int* m2 = (unsigned int*)alloc((size_t)NNODE * 4);
    float*    den2 = (float*)   alloc((size_t)NNODE * 4);
    float*    out2 = (float*)   alloc((size_t)NNODE * CCH * 4);
    float*    al2  = (float*)   alloc((size_t)ETOT * 4);
    unsigned int* pm = (unsigned int*)alloc((size_t)NGRAPH * CCH * 4);

    auto g = [](long n) { return (unsigned)((n + 255) / 256); };

    // init accumulators
    fill_f32<<<g((long)NNODE*HC), 256, 0, stream>>>(out1, NNODE*HC, 0.0f);
    fill_f32<<<g((long)NNODE*HEADS), 256, 0, stream>>>(den1, NNODE*HEADS, 0.0f);
    fill_f32<<<g(NNODE*CCH), 256, 0, stream>>>(out2, NNODE*CCH, 0.0f);
    fill_f32<<<g(NNODE), 256, 0, stream>>>(den2, NNODE, 0.0f);
    fill_u32<<<g((long)NNODE*HEADS), 256, 0, stream>>>(m1, NNODE*HEADS, 0u);
    fill_u32<<<g(NNODE), 256, 0, stream>>>(m2, NNODE, 0u);
    fill_u32<<<g(NGRAPH*CCH), 256, 0, stream>>>(pm, NGRAPH*CCH, 0x007FFFFFu); // map(-inf)

    // ---- layer 1 ----
    cast_pad_x<<<g((long)NNODE*KP1), 256, 0, stream>>>(x, xh);
    tcast_w1<<<g((long)HC*KP1), 256, 0, stream>>>(W1, w1t);
    {
        int groups = (NNODE/16) * (HC/64);    // 625 * 20
        wmma_gemm_f16<<<(groups + 7)/8, 256, 0, stream>>>(xh, w1t, h1, NNODE, HC, KP1);
    }
    attn_dot<<<g((long)NNODE*HEADS), 256, 0, stream>>>(h1, as1, asr1, HEADS);
    attn_dot<<<g((long)NNODE*HEADS), 256, 0, stream>>>(h1, ad1, ads1, HEADS);
    edge_score<<<g((long)ETOT*HEADS), 256, 0, stream>>>(ei, asr1, ads1, e1, m1, HEADS);
    decode_m<<<g((long)NNODE*HEADS), 256, 0, stream>>>(m1, NNODE*HEADS);
    edge_exp<<<g((long)ETOT*HEADS), 256, 0, stream>>>(ei, e1, (const float*)m1, den1, HEADS);
    edge_alpha<<<g((long)ETOT*HEADS), 256, 0, stream>>>(ei, e1, den1, alpha1_out, HEADS);
    aggregate<<<ETOT*HEADS, 128, 0, stream>>>(ei, h1, alpha1_out, out1, HEADS);
    elu_cast1<<<g((long)NNODE*HC), 256, 0, stream>>>(out1, b1, h1h);

    // ---- layer 2 ----
    tcast_w2<<<g((long)CCH*HC), 256, 0, stream>>>(W2, w2t);
    {
        int groups = (NNODE/16) * (CCH/64);   // 625 * 2
        wmma_gemm_f16<<<(groups + 7)/8, 256, 0, stream>>>(h1h, w2t, h2, NNODE, CCH, HC);
    }
    attn_dot<<<g(NNODE), 256, 0, stream>>>(h2, as2, asr2, 1);
    attn_dot<<<g(NNODE), 256, 0, stream>>>(h2, ad2, ads2, 1);
    edge_score<<<g(ETOT), 256, 0, stream>>>(ei, asr2, ads2, e2, m2, 1);
    decode_m<<<g(NNODE), 256, 0, stream>>>(m2, NNODE);
    edge_exp<<<g(ETOT), 256, 0, stream>>>(ei, e2, (const float*)m2, den2, 1);
    edge_alpha<<<g(ETOT), 256, 0, stream>>>(ei, e2, den2, al2, 1);
    aggregate<<<ETOT, 128, 0, stream>>>(ei, h2, al2, out2, 1);
    elu2_inplace<<<g(NNODE*CCH), 256, 0, stream>>>(out2, b2);

    // ---- pooling ----
    pool_max<<<g(NNODE*CCH), 256, 0, stream>>>(out2, bat, pm);
    decode_pool<<<g(NGRAPH*CCH), 256, 0, stream>>>(pm, out);
}
